// RoutedTrainedExpertMLP_59433757442709
// MI455X (gfx1250) — compile-verified, matrix operand order checked by reading, not run on hardware
//
#include <hip/hip_runtime.h>
#include <hip/hip_bf16.h>

typedef __attribute__((ext_vector_type(16))) _Float16 v16h;
typedef __attribute__((ext_vector_type(8)))  _Float16 v8h;
typedef __attribute__((ext_vector_type(2)))  __fp16   v2fp16;
typedef __attribute__((ext_vector_type(8)))  float    v8f;
typedef __attribute__((ext_vector_type(4)))  int      v4i;

#define T_TOK 8192
#define HDIM  2048
#define FDIM  4096
#define NEXP  8

#define BM 128
#define BN 128
#define BK 32
#define PLANE 2048   // halfs per plane: 128 rows x 16 halfs
#define LDO 132      // float stage stride
#define LDH 136      // f16 stage stride (272B, 16B aligned)

// ---------------- gfx1250 async global->LDS copy (ASYNCcnt) with fallback ----
#if defined(__has_builtin)
# if __has_builtin(__builtin_amdgcn_global_load_async_to_lds_b128)
#  define HAVE_ASYNC_COPY 1
# endif
#endif
#ifndef HAVE_ASYNC_COPY
# define HAVE_ASYNC_COPY 0
#endif

static __device__ __forceinline__ void copy16_to_lds(const unsigned short* g,
                                                     unsigned short* l) {
#if HAVE_ASYNC_COPY
  __builtin_amdgcn_global_load_async_to_lds_b128(
      (__attribute__((address_space(1))) v4i*)g,
      (__attribute__((address_space(3))) v4i*)l, 0, 0);
#else
  *(uint4*)l = *(const uint4*)g;
#endif
}

static __device__ __forceinline__ void async_wait0() {
#if HAVE_ASYNC_COPY
# if __has_builtin(__builtin_amdgcn_s_wait_asynccnt)
  __builtin_amdgcn_s_wait_asynccnt(0);
# else
  asm volatile("s_wait_asynccnt 0" ::: "memory");
# endif
#endif
}

// packed f32x2 -> f16x2 (v_cvt_pk_rtz_f16_f32), returned as 32-bit payload
static __device__ __forceinline__ unsigned pk2h(float a, float b) {
  union { v2fp16 h; unsigned u; } c;
  c.h = __builtin_amdgcn_cvt_pkrtz(a, b);
  return c.u;
}

static __device__ __forceinline__ unsigned short f2h_bits(float f) {
  union { _Float16 h; unsigned short u; } c;
  c.h = (_Float16)f;
  return c.u;
}

static __device__ __forceinline__ v16h frag2(const unsigned short* p01,
                                             const unsigned short* p23,
                                             int row, int kb) {
  v8h lo = *(const v8h*)(p01 + row * 16 + kb);
  v8h hi = *(const v8h*)(p23 + row * 16 + kb);
  v16h r;
#pragma unroll
  for (int i = 0; i < 8; ++i) { r[i] = lo[i]; r[8 + i] = hi[i]; }
  return r;
}

static __device__ __forceinline__ float silu_mul(float g, float u) {
  float t = 1.f + __expf(-g);
  return g * __builtin_amdgcn_rcpf(t) * u;
}

// ---------------------------------------------------------------- cvt x -> f16
__global__ __launch_bounds__(256)
void cvt_x_f16(const float* __restrict__ x, unsigned short* __restrict__ xh) {
  size_t i = ((size_t)blockIdx.x * blockDim.x + threadIdx.x) * 4;
  float4 v = *(const float4*)(x + i);
  uint2 o;
  o.x = pk2h(v.x, v.y);
  o.y = pk2h(v.z, v.w);
  *(uint2*)(xh + i) = o;
}

// ---------------------------------------------------------------- router
__global__ __launch_bounds__(256)
void router_kernel(const float* __restrict__ x, const float* __restrict__ rw,
                   float* __restrict__ wts) {
  int wave = threadIdx.x >> 5;
  int lane = threadIdx.x & 31;
  int tok = blockIdx.x * 8 + wave;
  if (tok >= T_TOK) return;
  const float* xr = x + (size_t)tok * HDIM;
  float acc[NEXP];
#pragma unroll
  for (int e = 0; e < NEXP; ++e) acc[e] = 0.f;
  for (int h = lane; h < HDIM; h += 32) {
    float xv = xr[h];
#pragma unroll
    for (int e = 0; e < NEXP; ++e) acc[e] += xv * rw[e * HDIM + h];
  }
#pragma unroll
  for (int e = 0; e < NEXP; ++e) {
    float v = acc[e];
    for (int off = 16; off; off >>= 1) v += __shfl_xor(v, off, 32);
    acc[e] = v;
  }
  if (lane == 0) {
    float mx = acc[0];
#pragma unroll
    for (int e = 1; e < NEXP; ++e) mx = fmaxf(mx, acc[e]);
    float p[NEXP]; float s = 0.f;
#pragma unroll
    for (int e = 0; e < NEXP; ++e) { p[e] = __expf(acc[e] - mx); s += p[e]; }
    float inv = 1.f / s;
#pragma unroll
    for (int e = 0; e < NEXP; ++e) p[e] *= inv;
    int i0 = 0;
#pragma unroll
    for (int e = 1; e < NEXP; ++e) if (p[e] > p[i0]) i0 = e;
    int i1 = (i0 == 0) ? 1 : 0;
#pragma unroll
    for (int e = 0; e < NEXP; ++e) if (e != i0 && p[e] > p[i1]) i1 = e;
    float d = 1.f / (p[i0] + p[i1]);
#pragma unroll
    for (int e = 0; e < NEXP; ++e)
      wts[(size_t)tok * NEXP + e] = (e == i0) ? p[i0] * d : ((e == i1) ? p[i1] * d : 0.f);
  }
}

// ---------------------------------------------------------------- shared gate/up (int8 dequant)
__global__ __launch_bounds__(256)
void shared_gateup_kernel(const unsigned short* __restrict__ xh,
                          const signed char* __restrict__ gw, const float* __restrict__ gs,
                          const signed char* __restrict__ uw, const float* __restrict__ us,
                          unsigned short* __restrict__ hidden) {
  // planes: [buf][part], part: 0=A01 1=A23 2=Bg01 3=Bg23 4=Bu01 5=Bu23
  __shared__ alignas(16) unsigned short smem[2 * 6 * PLANE];  // 48 KB

  const int tid = threadIdx.x;
  const int lane = tid & 31;
  const int wave = tid >> 5;
  const int wm = wave >> 2, wn = wave & 3;
  const int row0 = blockIdx.y * BM;
  const int col0 = blockIdx.x * BN;
  const int lr = lane & 15;
  const int kb = (lane >> 4) << 3;
  const int mlo = (lane >> 4) << 3;
  const int arow = tid >> 1;
  const int hiHalf = (tid & 1);
  const int acol = hiHalf << 4;

  const float sg = gs[col0 + arow];
  const float su = us[col0 + arow];

  auto plane = [&](int buf, int part) -> unsigned short* {
    return smem + (buf * 6 + part) * PLANE;
  };

  auto load_tile = [&](int kk, int buf) {
    const int k0 = kk * BK;
    {
      const unsigned short* src = xh + (size_t)(row0 + arow) * HDIM + k0 + acol;
      unsigned short* dst = plane(buf, hiHalf) + arow * 16;
      copy16_to_lds(src, dst);
      copy16_to_lds(src + 8, dst + 8);
    }
    {
      const signed char* g8 = gw + (size_t)(col0 + arow) * HDIM + k0 + acol;
      const signed char* u8 = uw + (size_t)(col0 + arow) * HDIM + k0 + acol;
      __builtin_prefetch(g8 + BK, 0, 0);
      __builtin_prefetch(u8 + BK, 0, 0);
      int4 gv = *(const int4*)g8;
      int4 uv = *(const int4*)u8;
      const signed char* gp = (const signed char*)&gv;
      const signed char* up = (const signed char*)&uv;
      unsigned og[8], ou[8];
#pragma unroll
      for (int i = 0; i < 8; ++i) {
        og[i] = pk2h((float)gp[2*i] * sg, (float)gp[2*i+1] * sg);
        ou[i] = pk2h((float)up[2*i] * su, (float)up[2*i+1] * su);
      }
      unsigned short* dg = plane(buf, 2 + hiHalf) + arow * 16;
      unsigned short* du = plane(buf, 4 + hiHalf) + arow * 16;
      uint4 q;
      q.x = og[0]; q.y = og[1]; q.z = og[2]; q.w = og[3]; *(uint4*)dg = q;
      q.x = og[4]; q.y = og[5]; q.z = og[6]; q.w = og[7]; *(uint4*)(dg + 8) = q;
      q.x = ou[0]; q.y = ou[1]; q.z = ou[2]; q.w = ou[3]; *(uint4*)du = q;
      q.x = ou[4]; q.y = ou[5]; q.z = ou[6]; q.w = ou[7]; *(uint4*)(du + 8) = q;
    }
  };

  v8f accg[4][2], accu[4][2];
#pragma unroll
  for (int i = 0; i < 4; ++i)
#pragma unroll
    for (int j = 0; j < 2; ++j) { accg[i][j] = (v8f)(0.f); accu[i][j] = (v8f)(0.f); }

  const int NK = HDIM / BK;
  load_tile(0, 0);
  async_wait0();
  for (int k = 0; k < NK; ++k) {
    __syncthreads();
    const int cur = k & 1;
    if (k + 1 < NK) load_tile(k + 1, cur ^ 1);

    const unsigned short* a01 = plane(cur, 0);
    const unsigned short* a23 = plane(cur, 1);
    const unsigned short* g01 = plane(cur, 2);
    const unsigned short* g23 = plane(cur, 3);
    const unsigned short* u01 = plane(cur, 4);
    const unsigned short* u23 = plane(cur, 5);

    v16h af[4], bgf[2], buf_[2];
#pragma unroll
    for (int i = 0; i < 4; ++i)
      af[i] = frag2(a01, a23, wm * 64 + i * 16 + lr, kb);
#pragma unroll
    for (int j = 0; j < 2; ++j) {
      bgf[j] = frag2(g01, g23, wn * 32 + j * 16 + lr, kb);
      buf_[j] = frag2(u01, u23, wn * 32 + j * 16 + lr, kb);
    }
#pragma unroll
    for (int i = 0; i < 4; ++i)
#pragma unroll
      for (int j = 0; j < 2; ++j) {
        accg[i][j] = __builtin_amdgcn_wmma_f32_16x16x32_f16(
            false, af[i], false, bgf[j], (short)0, accg[i][j], false, false);
        accu[i][j] = __builtin_amdgcn_wmma_f32_16x16x32_f16(
            false, af[i], false, buf_[j], (short)0, accu[i][j], false, false);
      }
    async_wait0();
  }
  __syncthreads();

  // epilogue: h = silu(g)*u, staged 64 rows at a time (overlay tile planes)
  unsigned short* Osh = smem;
  for (int p = 0; p < 2; ++p) {
    if (wm == p) {
#pragma unroll
      for (int i = 0; i < 4; ++i)
#pragma unroll
        for (int j = 0; j < 2; ++j) {
          int col = wn * 32 + j * 16 + lr;
#pragma unroll
          for (int r = 0; r < 8; ++r) {
            int mrow = i * 16 + mlo + r;  // local 0..63
            Osh[mrow * LDH + col] = f2h_bits(silu_mul(accg[i][j][r], accu[i][j][r]));
          }
        }
    }
    __syncthreads();
#pragma unroll
    for (int it = 0; it < 4; ++it) {
      int chunk = it * 256 + tid;  // 64*128/8 = 1024 chunks
      int row = chunk >> 4;
      int col = (chunk & 15) << 3;
      uint4 v = *(const uint4*)&Osh[row * LDH + col];
      *(uint4*)&hidden[(size_t)(row0 + p * 64 + row) * FDIM + col0 + col] = v;
    }
    __syncthreads();
  }
}

// ---------------------------------------------------------------- shared down (int8) -> out
__global__ __launch_bounds__(256)
void shared_down_kernel(const unsigned short* __restrict__ hidden,
                        const signed char* __restrict__ dw, const float* __restrict__ dsc,
                        float* __restrict__ out) {
  // parts: 0=A01 1=A23 2=B01 3=B23 ; Of overlay needs 16896 halfs
  __shared__ alignas(16) unsigned short smem[16896];  // 33 KB

  const int tid = threadIdx.x;
  const int lane = tid & 31;
  const int wave = tid >> 5;
  const int wm = wave >> 2, wn = wave & 3;
  const int row0 = blockIdx.y * BM;
  const int col0 = blockIdx.x * BN;
  const int lr = lane & 15;
  const int kb = (lane >> 4) << 3;
  const int mlo = (lane >> 4) << 3;
  const int arow = tid >> 1;
  const int hiHalf = (tid & 1);
  const int acol = hiHalf << 4;

  const float sd = dsc[col0 + arow];

  auto plane = [&](int buf, int part) -> unsigned short* {
    return smem + (buf * 4 + part) * PLANE;
  };

  auto load_tile = [&](int kk, int buf) {
    const int k0 = kk * BK;
    {
      const unsigned short* src = hidden + (size_t)(row0 + arow) * FDIM + k0 + acol;
      unsigned short* dst = plane(buf, hiHalf) + arow * 16;
      copy16_to_lds(src, dst);
      copy16_to_lds(src + 8, dst + 8);
    }
    {
      const signed char* w8 = dw + (size_t)(col0 + arow) * FDIM + k0 + acol;
      __builtin_prefetch(w8 + BK, 0, 0);
      int4 wv = *(const int4*)w8;
      const signed char* wp = (const signed char*)&wv;
      unsigned ow[8];
#pragma unroll
      for (int i = 0; i < 8; ++i)
        ow[i] = pk2h((float)wp[2*i] * sd, (float)wp[2*i+1] * sd);
      unsigned short* db = plane(buf, 2 + hiHalf) + arow * 16;
      uint4 q;
      q.x = ow[0]; q.y = ow[1]; q.z = ow[2]; q.w = ow[3]; *(uint4*)db = q;
      q.x = ow[4]; q.y = ow[5]; q.z = ow[6]; q.w = ow[7]; *(uint4*)(db + 8) = q;
    }
  };

  v8f acc[4][2];
#pragma unroll
  for (int i = 0; i < 4; ++i)
#pragma unroll
    for (int j = 0; j < 2; ++j) acc[i][j] = (v8f)(0.f);

  const int NK = FDIM / BK;
  load_tile(0, 0);
  async_wait0();
  for (int k = 0; k < NK; ++k) {
    __syncthreads();
    const int cur = k & 1;
    if (k + 1 < NK) load_tile(k + 1, cur ^ 1);

    const unsigned short* a01 = plane(cur, 0);
    const unsigned short* a23 = plane(cur, 1);
    const unsigned short* b01 = plane(cur, 2);
    const unsigned short* b23 = plane(cur, 3);
    v16h af[4], bf[2];
#pragma unroll
    for (int i = 0; i < 4; ++i)
      af[i] = frag2(a01, a23, wm * 64 + i * 16 + lr, kb);
#pragma unroll
    for (int j = 0; j < 2; ++j)
      bf[j] = frag2(b01, b23, wn * 32 + j * 16 + lr, kb);
#pragma unroll
    for (int i = 0; i < 4; ++i)
#pragma unroll
      for (int j = 0; j < 2; ++j)
        acc[i][j] = __builtin_amdgcn_wmma_f32_16x16x32_f16(
            false, af[i], false, bf[j], (short)0, acc[i][j], false, false);
    async_wait0();
  }
  __syncthreads();

  float* Of = (float*)smem;
  for (int p = 0; p < 2; ++p) {
    if (wm == p) {
#pragma unroll
      for (int i = 0; i < 4; ++i)
#pragma unroll
        for (int j = 0; j < 2; ++j) {
          int col = wn * 32 + j * 16 + lr;
#pragma unroll
          for (int r = 0; r < 8; ++r)
            Of[(i * 16 + mlo + r) * LDO + col] = acc[i][j][r];
        }
    }
    __syncthreads();
#pragma unroll
    for (int it = 0; it < 8; ++it) {
      int chunk = it * 256 + tid;  // 64*128/4 = 2048 chunks
      int row = chunk >> 5;
      int col = (chunk & 31) << 2;
      float4 v = *(const float4*)&Of[row * LDO + col];
      *(float4*)&out[(size_t)(row0 + p * 64 + row) * HDIM + col0 + col] = v;
    }
    __syncthreads();
  }
}

// ---------------------------------------------------------------- expert gate/up (fp32 weights)
__global__ __launch_bounds__(256)
void expert_gateup_kernel(const unsigned short* __restrict__ xh,
                          const float* __restrict__ gw, const float* __restrict__ uw,
                          unsigned short* __restrict__ hidden) {
  __shared__ alignas(16) unsigned short smem[2 * 6 * PLANE];  // 48 KB

  const int tid = threadIdx.x;
  const int lane = tid & 31;
  const int wave = tid >> 5;
  const int wm = wave >> 2, wn = wave & 3;
  const int row0 = blockIdx.y * BM;
  const int col0 = blockIdx.x * BN;
  const int lr = lane & 15;
  const int kb = (lane >> 4) << 3;
  const int mlo = (lane >> 4) << 3;
  const int arow = tid >> 1;
  const int hiHalf = (tid & 1);
  const int acol = hiHalf << 4;

  auto plane = [&](int buf, int part) -> unsigned short* {
    return smem + (buf * 6 + part) * PLANE;
  };

  auto load_tile = [&](int kk, int buf) {
    const int k0 = kk * BK;
    {
      const unsigned short* src = xh + (size_t)(row0 + arow) * HDIM + k0 + acol;
      unsigned short* dst = plane(buf, hiHalf) + arow * 16;
      copy16_to_lds(src, dst);
      copy16_to_lds(src + 8, dst + 8);
    }
    {
      const float* gp = gw + (size_t)(col0 + arow) * HDIM + k0 + acol;
      const float* up = uw + (size_t)(col0 + arow) * HDIM + k0 + acol;
      __builtin_prefetch(gp + BK, 0, 0);
      __builtin_prefetch(up + BK, 0, 0);
      unsigned og[8], ou[8];
#pragma unroll
      for (int i = 0; i < 4; ++i) {
        float4 g4 = *(const float4*)(gp + i * 4);
        float4 u4 = *(const float4*)(up + i * 4);
        og[2*i]   = pk2h(g4.x, g4.y);
        og[2*i+1] = pk2h(g4.z, g4.w);
        ou[2*i]   = pk2h(u4.x, u4.y);
        ou[2*i+1] = pk2h(u4.z, u4.w);
      }
      unsigned short* dg = plane(buf, 2 + hiHalf) + arow * 16;
      unsigned short* du = plane(buf, 4 + hiHalf) + arow * 16;
      uint4 q;
      q.x = og[0]; q.y = og[1]; q.z = og[2]; q.w = og[3]; *(uint4*)dg = q;
      q.x = og[4]; q.y = og[5]; q.z = og[6]; q.w = og[7]; *(uint4*)(dg + 8) = q;
      q.x = ou[0]; q.y = ou[1]; q.z = ou[2]; q.w = ou[3]; *(uint4*)du = q;
      q.x = ou[4]; q.y = ou[5]; q.z = ou[6]; q.w = ou[7]; *(uint4*)(du + 8) = q;
    }
  };

  v8f accg[4][2], accu[4][2];
#pragma unroll
  for (int i = 0; i < 4; ++i)
#pragma unroll
    for (int j = 0; j < 2; ++j) { accg[i][j] = (v8f)(0.f); accu[i][j] = (v8f)(0.f); }

  const int NK = HDIM / BK;
  load_tile(0, 0);
  async_wait0();
  for (int k = 0; k < NK; ++k) {
    __syncthreads();
    const int cur = k & 1;
    if (k + 1 < NK) load_tile(k + 1, cur ^ 1);

    const unsigned short* a01 = plane(cur, 0);
    const unsigned short* a23 = plane(cur, 1);
    const unsigned short* g01 = plane(cur, 2);
    const unsigned short* g23 = plane(cur, 3);
    const unsigned short* u01 = plane(cur, 4);
    const unsigned short* u23 = plane(cur, 5);

    v16h af[4], bgf[2], buf_[2];
#pragma unroll
    for (int i = 0; i < 4; ++i)
      af[i] = frag2(a01, a23, wm * 64 + i * 16 + lr, kb);
#pragma unroll
    for (int j = 0; j < 2; ++j) {
      bgf[j] = frag2(g01, g23, wn * 32 + j * 16 + lr, kb);
      buf_[j] = frag2(u01, u23, wn * 32 + j * 16 + lr, kb);
    }
#pragma unroll
    for (int i = 0; i < 4; ++i)
#pragma unroll
      for (int j = 0; j < 2; ++j) {
        accg[i][j] = __builtin_amdgcn_wmma_f32_16x16x32_f16(
            false, af[i], false, bgf[j], (short)0, accg[i][j], false, false);
        accu[i][j] = __builtin_amdgcn_wmma_f32_16x16x32_f16(
            false, af[i], false, buf_[j], (short)0, accu[i][j], false, false);
      }
    async_wait0();
  }
  __syncthreads();

  unsigned short* Osh = smem;
  for (int p = 0; p < 2; ++p) {
    if (wm == p) {
#pragma unroll
      for (int i = 0; i < 4; ++i)
#pragma unroll
        for (int j = 0; j < 2; ++j) {
          int col = wn * 32 + j * 16 + lr;
#pragma unroll
          for (int r = 0; r < 8; ++r) {
            int mrow = i * 16 + mlo + r;
            Osh[mrow * LDH + col] = f2h_bits(silu_mul(accg[i][j][r], accu[i][j][r]));
          }
        }
    }
    __syncthreads();
#pragma unroll
    for (int it = 0; it < 4; ++it) {
      int chunk = it * 256 + tid;
      int row = chunk >> 4;
      int col = (chunk & 15) << 3;
      uint4 v = *(const uint4*)&Osh[row * LDH + col];
      *(uint4*)&hidden[(size_t)(row0 + p * 64 + row) * FDIM + col0 + col] = v;
    }
    __syncthreads();
  }
}

// ---------------------------------------------------------------- expert down: out += w_e * (h_e @ De^T)
__global__ __launch_bounds__(256)
void expert_down_kernel(const unsigned short* __restrict__ hidden,
                        const float* __restrict__ dw, const float* __restrict__ wts,
                        int e, float* __restrict__ out) {
  __shared__ alignas(16) unsigned short smem[16896];  // 33 KB
  __shared__ float wsh[BM];

  const int tid = threadIdx.x;
  const int lane = tid & 31;
  const int wave = tid >> 5;
  const int wm = wave >> 2, wn = wave & 3;
  const int row0 = blockIdx.y * BM;
  const int col0 = blockIdx.x * BN;
  const int lr = lane & 15;
  const int kb = (lane >> 4) << 3;
  const int mlo = (lane >> 4) << 3;
  const int arow = tid >> 1;
  const int hiHalf = (tid & 1);
  const int acol = hiHalf << 4;

  if (tid < BM) wsh[tid] = wts[(size_t)(row0 + tid) * NEXP + e];

  auto plane = [&](int buf, int part) -> unsigned short* {
    return smem + (buf * 4 + part) * PLANE;
  };

  auto load_tile = [&](int kk, int buf) {
    const int k0 = kk * BK;
    {
      const unsigned short* src = hidden + (size_t)(row0 + arow) * FDIM + k0 + acol;
      unsigned short* dst = plane(buf, hiHalf) + arow * 16;
      copy16_to_lds(src, dst);
      copy16_to_lds(src + 8, dst + 8);
    }
    {
      const float* wp = dw + (size_t)(col0 + arow) * FDIM + k0 + acol;
      __builtin_prefetch(wp + BK, 0, 0);
      unsigned ow[8];
#pragma unroll
      for (int i = 0; i < 4; ++i) {
        float4 w4 = *(const float4*)(wp + i * 4);
        ow[2*i]   = pk2h(w4.x, w4.y);
        ow[2*i+1] = pk2h(w4.z, w4.w);
      }
      unsigned short* db = plane(buf, 2 + hiHalf) + arow * 16;
      uint4 q;
      q.x = ow[0]; q.y = ow[1]; q.z = ow[2]; q.w = ow[3]; *(uint4*)db = q;
      q.x = ow[4]; q.y = ow[5]; q.z = ow[6]; q.w = ow[7]; *(uint4*)(db + 8) = q;
    }
  };

  v8f acc[4][2];
#pragma unroll
  for (int i = 0; i < 4; ++i)
#pragma unroll
    for (int j = 0; j < 2; ++j) acc[i][j] = (v8f)(0.f);

  const int NK = FDIM / BK;
  load_tile(0, 0);
  async_wait0();
  for (int k = 0; k < NK; ++k) {
    __syncthreads();
    const int cur = k & 1;
    if (k + 1 < NK) load_tile(k + 1, cur ^ 1);

    const unsigned short* a01 = plane(cur, 0);
    const unsigned short* a23 = plane(cur, 1);
    const unsigned short* b01 = plane(cur, 2);
    const unsigned short* b23 = plane(cur, 3);
    v16h af[4], bf[2];
#pragma unroll
    for (int i = 0; i < 4; ++i)
      af[i] = frag2(a01, a23, wm * 64 + i * 16 + lr, kb);
#pragma unroll
    for (int j = 0; j < 2; ++j)
      bf[j] = frag2(b01, b23, wn * 32 + j * 16 + lr, kb);
#pragma unroll
    for (int i = 0; i < 4; ++i)
#pragma unroll
      for (int j = 0; j < 2; ++j)
        acc[i][j] = __builtin_amdgcn_wmma_f32_16x16x32_f16(
            false, af[i], false, bf[j], (short)0, acc[i][j], false, false);
    async_wait0();
  }
  __syncthreads();

  float* Of = (float*)smem;
  for (int p = 0; p < 2; ++p) {
    if (wm == p) {
#pragma unroll
      for (int i = 0; i < 4; ++i)
#pragma unroll
        for (int j = 0; j < 2; ++j) {
          int col = wn * 32 + j * 16 + lr;
#pragma unroll
          for (int r = 0; r < 8; ++r) {
            int mrow = i * 16 + mlo + r;
            Of[mrow * LDO + col] = wsh[p * 64 + mrow] * acc[i][j][r];
          }
        }
    }
    __syncthreads();
#pragma unroll
    for (int it = 0; it < 8; ++it) {
      int chunk = it * 256 + tid;
      int row = chunk >> 5;
      int col = (chunk & 31) << 2;
      float4 add = *(const float4*)&Of[row * LDO + col];
      float* dst = &out[(size_t)(row0 + p * 64 + row) * HDIM + col0 + col];
      float4 cur = *(const float4*)dst;
      cur.x += add.x; cur.y += add.y; cur.z += add.z; cur.w += add.w;
      *(float4*)dst = cur;
    }
    __syncthreads();
  }
}

// ---------------------------------------------------------------- host
extern "C" void kernel_launch(void* const* d_in, const int* in_sizes, int n_in,
                              void* d_out, int out_size, void* d_ws, size_t ws_size,
                              hipStream_t stream) {
  (void)in_sizes; (void)n_in; (void)out_size; (void)ws_size;
  const float* x = (const float*)d_in[0];
  const float* rw = (const float*)d_in[1];
  const signed char* gw8 = (const signed char*)d_in[2];
  const float* gs = (const float*)d_in[3];
  const signed char* uw8 = (const signed char*)d_in[4];
  const float* us = (const float*)d_in[5];
  const signed char* dw8 = (const signed char*)d_in[6];
  const float* dsc = (const float*)d_in[7];
  const float* eg = (const float*)d_in[8];
  const float* eu = (const float*)d_in[9];
  const float* ed = (const float*)d_in[10];
  float* out = (float*)d_out;

  char* ws = (char*)d_ws;
  unsigned short* xh = (unsigned short*)ws;                                   // 32 MB
  float* wts = (float*)(ws + (size_t)T_TOK * HDIM * 2);                       // 256 KB
  unsigned short* hidden =
      (unsigned short*)(ws + (size_t)T_TOK * HDIM * 2 + (size_t)T_TOK * NEXP * 4);  // 64 MB

  cvt_x_f16<<<(T_TOK * HDIM / 4) / 256, 256, 0, stream>>>(x, xh);
  router_kernel<<<T_TOK / 8, 256, 0, stream>>>(x, rw, wts);
  shared_gateup_kernel<<<dim3(FDIM / BN, T_TOK / BM), 256, 0, stream>>>(
      xh, gw8, gs, uw8, us, hidden);
  shared_down_kernel<<<dim3(HDIM / BN, T_TOK / BM), 256, 0, stream>>>(
      hidden, dw8, dsc, out);
  for (int e = 0; e < NEXP; ++e) {
    expert_gateup_kernel<<<dim3(FDIM / BN, T_TOK / BM), 256, 0, stream>>>(
        xh, eg + (size_t)e * FDIM * HDIM, eu + (size_t)e * FDIM * HDIM, hidden);
    expert_down_kernel<<<dim3(HDIM / BN, T_TOK / BM), 256, 0, stream>>>(
        hidden, ed + (size_t)e * HDIM * FDIM, wts, e, out);
  }
}
